// GCN_68066641707024
// MI455X (gfx1250) — matrix operand order repaired and checked by reading, not compiled, not run on hardware
//
#include <hip/hip_runtime.h>
#include <cstddef>

// ---------------------------------------------------------------------------
// Types / helpers
// ---------------------------------------------------------------------------
typedef __attribute__((ext_vector_type(4)))  float  f32x4;
typedef __attribute__((ext_vector_type(8)))  float  f32x8;
typedef __attribute__((ext_vector_type(8)))  __bf16 bf16x8;
typedef __attribute__((ext_vector_type(16))) __bf16 bf16x16;

template<bool C, typename A, typename B> struct Cond { using type = A; };
template<typename A, typename B> struct Cond<false, A, B> { using type = B; };

// Problem constants (from reference setup_inputs)
static constexpr int B_  = 16;
static constexpr int NV  = 1024;
static constexpr int NE  = 2048;
static constexpr int FV  = 128;
static constexpr int FE  = 64;
static constexpr int H_  = 128;
static constexpr int C_  = 16;

__device__ __forceinline__ void wait_async_zero() {
#if __has_builtin(__builtin_amdgcn_s_wait_asynccnt)
  __builtin_amdgcn_s_wait_asynccnt(0);
#else
  asm volatile("s_wait_asynccnt 0x0" ::: "memory");
#endif
}

// ---------------------------------------------------------------------------
// Fragment loads from LDS tiles stored [row][K], row stride LDA (bf16).
// A 16x32 bf16: lane m=lane%16; laneHi -> K {0..7,16..23} vs {8..15,24..31}:
//   two 16B chunks at k = ks+lhi*8 and ks+16+lhi*8.
// B 32x16 bf16: lane n=lane%16; same K order as A:
//   two 16B chunks at k = ks+lhi*8 and ks+16+lhi*8.
// ---------------------------------------------------------------------------
__device__ __forceinline__ bf16x16 load_fragA(const __bf16* p) {
  bf16x8 lo = *reinterpret_cast<const bf16x8*>(p);
  bf16x8 hi = *reinterpret_cast<const bf16x8*>(p + 16);
  bf16x16 r;
#pragma unroll
  for (int q = 0; q < 8; ++q) { r[q] = lo[q]; r[q + 8] = hi[q]; }
  return r;
}
__device__ __forceinline__ bf16x16 load_fragB(const __bf16* p) {
  return *reinterpret_cast<const bf16x16*>(p);
}

// ---------------------------------------------------------------------------
// Register-staged tile copy: load() issues global vector loads (data + scale,
// no consume -> no waits), store() converts (+relu, +per-k scale) to bf16 and
// writes the LDS tile branch-free.
// KN=false: global row-major BR x K.  KN=true: global K x BR.
// With NT % (BK/V) == 0 each thread's chunks share one fixed k-subrange, so
// the per-k scale is a single f32x4 (x2 for V=8) register preload.
// ---------------------------------------------------------------------------
template<typename T, bool KN, bool SCALED, bool RELU,
         int BR, int BKt, int NTt, int LDAt>
struct Stage {
  static constexpr int V   = (sizeof(T) == 4) ? 4 : 8;
  static constexpr int CPT = (BR * BKt) / (V * NTt);  // chunks per thread
  using VecT = typename Cond<sizeof(T) == 4, f32x4, bf16x8>::type;
  static_assert(!(SCALED && !KN) || (NTt % (BKt / V) == 0),
                "scale preload requires fixed k-subrange per thread");
  VecT r[CPT];
  alignas(16) float scv[(SCALED && !KN) ? V : 4];  // per-k scale (fixed kc)
  float sck[(SCALED && KN) ? CPT : 1];             // per-chunk scale (KN)

  __device__ __forceinline__ void load(const T* __restrict__ g, int base,
                                       int ld, int k0,
                                       const float* __restrict__ scaleK,
                                       int tid) {
    if constexpr (!KN) {
      constexpr int CPR = BKt / V;
      if constexpr (SCALED) {
        const int kc0 = (tid % CPR) * V;
        const float* sp = scaleK + k0 + kc0;
#pragma unroll
        for (int q = 0; q < V; q += 4)
          *reinterpret_cast<f32x4*>(&scv[q]) =
              *reinterpret_cast<const f32x4*>(sp + q);
      }
#pragma unroll
      for (int t = 0; t < CPT; ++t) {
        const int c = tid + t * NTt;
        const int rr = c / CPR, kc = (c % CPR) * V;
        r[t] = *reinterpret_cast<const VecT*>(g + (size_t)(base + rr) * ld + (k0 + kc));
      }
    } else {
      constexpr int CPR = BR / V;
#pragma unroll
      for (int t = 0; t < CPT; ++t) {
        const int c = tid + t * NTt;
        const int k = c / CPR, rc = (c % CPR) * V;
        if constexpr (SCALED) sck[t] = scaleK[k0 + k];
        r[t] = *reinterpret_cast<const VecT*>(g + (size_t)(k0 + k) * ld + (base + rc));
      }
    }
  }

  __device__ __forceinline__ void store(__bf16* lds, int tid) {
#pragma unroll
    for (int t = 0; t < CPT; ++t) {
      const int c = tid + t * NTt;
      if constexpr (!KN) {
        constexpr int CPR = BKt / V;
        const int rr = c / CPR, kc = (c % CPR) * V;
#pragma unroll
        for (int q = 0; q < V; ++q) {
          float x = (float)r[t][q];
          if constexpr (RELU)   x = fmaxf(x, 0.0f);
          if constexpr (SCALED) x *= scv[q];
          lds[rr * LDAt + kc + q] = (__bf16)x;
        }
      } else {
        constexpr int CPR = BR / V;
        const int k = c / CPR, rc = (c % CPR) * V;
#pragma unroll
        for (int q = 0; q < V; ++q) {
          float x = (float)r[t][q];
          if constexpr (RELU)   x = fmaxf(x, 0.0f);
          if constexpr (SCALED) x *= sck[t];
          lds[(rc + q) * LDAt + k] = (__bf16)x;
        }
      }
    }
  }
};

// ---------------------------------------------------------------------------
// Async global->LDS staging (bf16, no scale, row-major BR x K only):
// one GLOBAL_LOAD_ASYNC_TO_LDS_B128 per 16B chunk, tracked by ASYNCcnt.
// ---------------------------------------------------------------------------
template<int BR, int BKt, int NTt, int LDAt>
__device__ __forceinline__ void stage_async_bf16(__bf16* lds, const __bf16* __restrict__ g,
                                                 int base, int ld, int k0, int tid) {
  constexpr int V   = 8;
  constexpr int CPR = BKt / V;
  constexpr int CPT = (BR * BKt) / (V * NTt);
#pragma unroll
  for (int t = 0; t < CPT; ++t) {
    const int c  = tid + t * NTt;
    const int rr = c / CPR, kc = (c % CPR) * V;
    const __bf16* src = g + (size_t)(base + rr) * ld + (k0 + kc);
    const unsigned ldsOff = (unsigned)(size_t)(lds + rr * LDAt + kc);
    asm volatile("global_load_async_to_lds_b128 %0, %1, off"
                 :: "v"(ldsOff), "v"(src)
                 : "memory");
  }
}

// ---------------------------------------------------------------------------
// Double-buffered WMMA bf16 GEMM:  C[M,N] = opA(A) * opB(B)  (f32 accumulate)
//   A: row-major M x K (f32 or bf16), optional per-k scale, optional relu.
//   B: BKN=false -> row-major N x K ; BKN=true -> row-major K x N.
//   ASYNC_A/B: operand is bf16, unscaled -> global_load_async_to_lds_b128.
//   EPI=0: out_bf16[i,j] = bf16( (i==j ? 1 : acc) * adj[i,j] )
//   EPI=1: f = acc + bias[j]; relu?; f32 store (optionally transposed)
// blockDim = WGM*WGN*32 (wave32); wave tile = (BM/WGM) x (BN/WGN).
// ---------------------------------------------------------------------------
template<int BM, int BN, int BK, int WGM, int WGN, bool BKN,
         typename TA, typename TB, int EPI,
         bool ASYNC_A, bool ASYNC_B, bool SCA, bool SCB, bool RELUA>
__global__ __launch_bounds__(WGM* WGN * 32) void gemm_wmma_bf16(
    const TA* __restrict__ Ag, size_t strideAb, int lda,
    const TB* __restrict__ Bg, size_t strideBb, int ldb,
    const float* __restrict__ scaleA,
    const float* __restrict__ scaleB, size_t strideScb,
    const float* __restrict__ adj, size_t strideAdjb,
    __bf16* __restrict__ outBf, size_t strideObf,
    const float* __restrict__ bias,
    float* __restrict__ outF, size_t strideOf, int ldo, int reluOut,
    int transOut, int M, int N, int K) {
  constexpr int NT  = WGM * WGN * 32;
  constexpr int WMT = BM / (WGM * 16);
  constexpr int WNT = BN / (WGN * 16);
  constexpr int LDA = BK + 8;  // +16B row pad -> conflict-free b128 frag loads

  __shared__ __attribute__((aligned(16))) __bf16 sA[2 * BM * LDA];
  __shared__ __attribute__((aligned(16))) __bf16 sB[2 * BN * LDA];

  const int b  = blockIdx.z;
  const TA* A  = Ag + (size_t)b * strideAb;
  const TB* Bp = Bg + (size_t)b * strideBb;
  const float* scA = SCA ? scaleA + (size_t)b * strideScb : nullptr;
  const float* scB = SCB ? scaleB + (size_t)b * strideScb : nullptr;

  const int tid  = threadIdx.x;
  const int wave = tid >> 5;
  const int lane = tid & 31;
  const int lm   = lane & 15;
  const int lhi  = lane >> 4;
  const int wm   = wave / WGN;
  const int wn   = wave % WGN;
  const int rowBase = blockIdx.y * BM;
  const int colBase = blockIdx.x * BN;

  Stage<TA, false, SCA, RELUA, BM, BK, NT, LDA> stA;
  Stage<TB, BKN,   SCB, false, BN, BK, NT, LDA> stB;

  f32x8 acc[WMT][WNT] = {};

  // ---- prologue: fill buffer 0 ----
  if constexpr (ASYNC_A) stage_async_bf16<BM, BK, NT, LDA>(sA, A, rowBase, lda, 0, tid);
  else                   stA.load(A, rowBase, lda, 0, scA, tid);
  if constexpr (ASYNC_B) stage_async_bf16<BN, BK, NT, LDA>(sB, Bp, colBase, ldb, 0, tid);
  else                   stB.load(Bp, colBase, ldb, 0, scB, tid);
  if constexpr (!ASYNC_A) stA.store(sA, tid);
  if constexpr (!ASYNC_B) stB.store(sB, tid);
  if constexpr (ASYNC_A || ASYNC_B) wait_async_zero();
  __syncthreads();

  const int nsteps = K / BK;
  for (int i = 0; i < nsteps; ++i) {
    const int k0 = i * BK;
    __bf16* curA = sA + (i & 1) * (BM * LDA);
    __bf16* curB = sB + (i & 1) * (BN * LDA);
    __bf16* nxtA = sA + ((i + 1) & 1) * (BM * LDA);
    __bf16* nxtB = sB + ((i + 1) & 1) * (BN * LDA);
    const bool more = (i + 1) < nsteps;

    // Issue next-tile global traffic first (async ops / loads w/o consume)...
    if (more) {
      if constexpr (ASYNC_A) stage_async_bf16<BM, BK, NT, LDA>(nxtA, A, rowBase, lda, k0 + BK, tid);
      else                   stA.load(A, rowBase, lda, k0 + BK, scA, tid);
      if constexpr (ASYNC_B) stage_async_bf16<BN, BK, NT, LDA>(nxtB, Bp, colBase, ldb, k0 + BK, tid);
      else                   stB.load(Bp, colBase, ldb, k0 + BK, scB, tid);
    }

    // ...then run WMMA on the current LDS tiles while they are in flight.
#pragma unroll
    for (int ks = 0; ks < BK; ks += 32) {
      bf16x16 af[WMT];
      bf16x16 bfr[WNT];
#pragma unroll
      for (int ii = 0; ii < WMT; ++ii)
        af[ii] = load_fragA(&curA[(size_t)((wm * WMT + ii) * 16 + lm) * LDA + ks + lhi * 8]);
#pragma unroll
      for (int jj = 0; jj < WNT; ++jj)
        bfr[jj] = load_fragA(&curB[(size_t)((wn * WNT + jj) * 16 + lm) * LDA + ks + lhi * 8]);
#pragma unroll
      for (int ii = 0; ii < WMT; ++ii)
#pragma unroll
        for (int jj = 0; jj < WNT; ++jj)
          acc[ii][jj] = __builtin_amdgcn_wmma_f32_16x16x32_bf16(
              false, af[ii], false, bfr[jj], (short)0, acc[ii][jj], false, false);
    }

    if (more) {
      if constexpr (!ASYNC_A) stA.store(nxtA, tid);
      if constexpr (!ASYNC_B) stB.store(nxtB, tid);
      if constexpr (ASYNC_A || ASYNC_B) wait_async_zero();
      __syncthreads();
    }
  }

  // Epilogue (C/D layout: vgpr r -> row = r + 8*lhi; col = lane%16)
#pragma unroll
  for (int ii = 0; ii < WMT; ++ii)
#pragma unroll
    for (int jj = 0; jj < WNT; ++jj)
#pragma unroll
      for (int r = 0; r < 8; ++r) {
        const int row = rowBase + (wm * WMT + ii) * 16 + lhi * 8 + r;
        const int col = colBase + (wn * WNT + jj) * 16 + lm;
        float v = acc[ii][jj][r];
        if constexpr (EPI == 0) {
          const float a = adj[(size_t)b * strideAdjb + (size_t)row * N + col];
          const float c = ((row == col) ? 1.0f : v) * a;
          outBf[(size_t)b * strideObf + (size_t)row * N + col] = (__bf16)c;
        } else {
          if (bias) v += bias[col];
          if (reluOut) v = fmaxf(v, 0.0f);
          float* of = outF + (size_t)b * strideOf;
          if (transOut) of[(size_t)col * ldo + row] = v;
          else          of[(size_t)row * ldo + col] = v;
        }
      }
}

// ---------------------------------------------------------------------------
// Tt[b][e][v] = bf16(T[b][v][e])   (LDS-tiled 32x32 transpose, coalesced)
// ---------------------------------------------------------------------------
__global__ __launch_bounds__(256) void transpose_to_bf16(
    const float* __restrict__ T, __bf16* __restrict__ Tt, int rows, int cols) {
  __shared__ float tile[32][33];
  const int b  = blockIdx.z;
  const float* Tb = T + (size_t)b * rows * cols;
  __bf16* Ob = Tt + (size_t)b * rows * cols;
  const int c0 = blockIdx.x * 32, r0 = blockIdx.y * 32;
  const int tx = threadIdx.x, ty = threadIdx.y;
#pragma unroll
  for (int i = 0; i < 32; i += 8)
    tile[ty + i][tx] = Tb[(size_t)(r0 + ty + i) * cols + (c0 + tx)];
  __syncthreads();
#pragma unroll
  for (int i = 0; i < 32; i += 8)
    Ob[(size_t)(c0 + ty + i) * rows + (r0 + tx)] = (__bf16)tile[tx][ty + i];
}

// ---------------------------------------------------------------------------
// d[b*2048 + r] = sum_f in[b][r][f] * p[f]      (one wave per row)
// ---------------------------------------------------------------------------
__global__ __launch_bounds__(128) void rowdot_k(
    const float* __restrict__ in, const float* __restrict__ p,
    float* __restrict__ d, int R, int F, int totalRows) {
  const int wavesPerBlock = blockDim.x >> 5;
  const int row  = blockIdx.x * wavesPerBlock + (threadIdx.x >> 5);
  const int lane = threadIdx.x & 31;
  if (row >= totalRows) return;
  const float* src = in + (size_t)row * F;
  float s = 0.0f;
  for (int f = lane; f < F; f += 32) s += src[f] * p[f];
#pragma unroll
  for (int o = 16; o > 0; o >>= 1) s += __shfl_xor(s, o, 32);
  if (lane == 0) {
    const int b = row / R, r = row % R;
    d[(size_t)b * 2048 + r] = s;
  }
}

// ---------------------------------------------------------------------------
// rc[b*2048 + j] = 1 / max_i A[b][i][j]   (A bf16 n x n; coalesced over lanes)
// ---------------------------------------------------------------------------
__global__ __launch_bounds__(256) void colmax_rcp_k(
    const __bf16* __restrict__ A, float* __restrict__ rc, int n) {
  const int j = blockIdx.x * blockDim.x + threadIdx.x;
  const int b = blockIdx.z;
  if (j >= n) return;
  const __bf16* Ab = A + (size_t)b * n * n;
  float m = -3.4e38f;
  for (int i = 0; i < n; ++i) m = fmaxf(m, (float)Ab[(size_t)i * n + j]);
  rc[(size_t)b * 2048 + j] = 1.0f / m;
}

// ---------------------------------------------------------------------------
// Host orchestration
// ---------------------------------------------------------------------------
extern "C" void kernel_launch(void* const* d_in, const int* in_sizes, int n_in,
                              void* d_out, int out_size, void* d_ws,
                              size_t ws_size, hipStream_t stream) {
  const float* X    = (const float*)d_in[0];
  const float* Z    = (const float*)d_in[1];
  const float* adjE = (const float*)d_in[2];
  const float* adjV = (const float*)d_in[3];
  const float* T    = (const float*)d_in[4];
  const float* W1   = (const float*)d_in[5];
  const float* b1   = (const float*)d_in[6];
  const float* p1   = (const float*)d_in[7];
  const float* W2   = (const float*)d_in[8];
  const float* b2   = (const float*)d_in[9];
  const float* p2   = (const float*)d_in[10];
  const float* W3   = (const float*)d_in[11];
  const float* b3   = (const float*)d_in[12];
  const float* p3   = (const float*)d_in[13];
  float* out = (float*)d_out;

  // Workspace layout (bytes): ~227 MB total
  char* w = (char*)d_ws;
  __bf16* Af  = (__bf16*)(w);                      // 16*2048*2048*2 = 128MB
  __bf16* Tt  = (__bf16*)(w + 134217728ull);       // 16*2048*1024*2 =  64MB
  float*  X1  = (float*)(w + 201326592ull);        // 16*1024*128*4  =   8MB
  float*  Z2  = (float*)(w + 209715200ull);        // 16*2048*64*4   =   8MB
  float*  XWT = (float*)(w + 218103808ull);        // up to            8MB
  float*  dbf = (float*)(w + 226492416ull);        // 16*2048*4      = 128KB
  float*  rcm = (float*)(w + 226623488ull);        // 16*2048*4      = 128KB
  (void)in_sizes; (void)n_in; (void)out_size; (void)ws_size;

  // ---- Precompute: Tt = bf16(T^T) per batch (for TN-shaped edge layer) ----
  transpose_to_bf16<<<dim3(NE / 32, NV / 32, B_), dim3(32, 8), 0, stream>>>(
      T, Tt, NV, NE);

  // ================= Layer 1 (node): d1 = Z @ p1 ==========================
  rowdot_k<<<(B_ * NE) / 4, 128, 0, stream>>>(Z, p1, dbf, NE, FE, B_ * NE);

  // mult1 = (T*diag(d1)) @ T^T ; fuse diag<-1, *adj_v ; store bf16 Af (1024^2)
  gemm_wmma_bf16<128, 128, 64, 2, 4, false, float, float, 0,
                 false, false, true, false, false>
      <<<dim3(NV / 128, NV / 128, B_), 256, 0, stream>>>(
          T, (size_t)NV * NE, NE, T, (size_t)NV * NE, NE,
          dbf, nullptr, 2048,
          adjV, (size_t)NV * NV, Af, (size_t)NV * NV,
          nullptr, nullptr, 0, 0, 0, 0, NV, NV, NE);

  colmax_rcp_k<<<dim3(NV / 256, 1, B_), 256, 0, stream>>>(Af, rcm, NV);

  // XW1^T = (X @ W1)^T  (stored 128 x 1024 f32 per batch)
  gemm_wmma_bf16<128, 128, 64, 2, 4, true, float, float, 1,
                 false, false, false, false, false>
      <<<dim3(H_ / 128, NV / 128, B_), 256, 0, stream>>>(
          X, (size_t)NV * FV, FV, W1, 0, H_,
          nullptr, nullptr, 0, nullptr, 0, nullptr, 0,
          nullptr, XWT, (size_t)H_ * NV, NV, 0, 1, NV, H_, FV);

  // X1 = relu( Af @ (XW1 * rcmax) + b1 )   (A staged via async-LDS)
  gemm_wmma_bf16<128, 128, 64, 2, 4, false, __bf16, float, 1,
                 true, false, false, true, false>
      <<<dim3(H_ / 128, NV / 128, B_), 256, 0, stream>>>(
          Af, (size_t)NV * NV, NV, XWT, (size_t)H_ * NV, NV,
          nullptr, rcm, 2048, nullptr, 0, nullptr, 0,
          b1, X1, (size_t)NV * H_, H_, 1, 0, NV, H_, NV);

  // ================= Layer 2 (edge): d2 = X1 @ p2 =========================
  rowdot_k<<<(B_ * NV) / 4, 128, 0, stream>>>(X1, p2, dbf, NV, H_, B_ * NV);

  // mult2 = (Tt*diag(d2)) @ Tt^T ; *adj_e ; Af 2048^2 (B staged via async-LDS)
  gemm_wmma_bf16<128, 128, 64, 2, 4, false, __bf16, __bf16, 0,
                 false, true, true, false, false>
      <<<dim3(NE / 128, NE / 128, B_), 256, 0, stream>>>(
          Tt, (size_t)NE * NV, NV, Tt, (size_t)NE * NV, NV,
          dbf, nullptr, 2048,
          adjE, (size_t)NE * NE, Af, (size_t)NE * NE,
          nullptr, nullptr, 0, 0, 0, 0, NE, NE, NV);

  colmax_rcp_k<<<dim3(NE / 256, 1, B_), 256, 0, stream>>>(Af, rcm, NE);

  // ZW2^T = (relu(Z) @ W2)^T  (64 x 2048 f32 per batch)
  gemm_wmma_bf16<128, 64, 64, 4, 2, true, float, float, 1,
                 false, false, false, false, true>
      <<<dim3(FE / 64, NE / 128, B_), 256, 0, stream>>>(
          Z, (size_t)NE * FE, FE, W2, 0, FE,
          nullptr, nullptr, 0, nullptr, 0, nullptr, 0,
          nullptr, XWT, (size_t)FE * NE, NE, 0, 1, NE, FE, FE);

  // Z2 = relu( Af @ (ZW2 * rcmax) + b2 )   (A staged via async-LDS)
  gemm_wmma_bf16<128, 64, 64, 4, 2, false, __bf16, float, 1,
                 true, false, false, true, false>
      <<<dim3(FE / 64, NE / 128, B_), 256, 0, stream>>>(
          Af, (size_t)NE * NE, NE, XWT, (size_t)FE * NE, NE,
          nullptr, rcm, 2048, nullptr, 0, nullptr, 0,
          b2, Z2, (size_t)NE * FE, FE, 1, 0, NE, FE, NE);

  // ================= Layer 3 (node): d3 = Z2 @ p3 =========================
  rowdot_k<<<(B_ * NE) / 4, 128, 0, stream>>>(Z2, p3, dbf, NE, FE, B_ * NE);

  // mult3 = (T*diag(d3)) @ T^T ; *adj_v ; Af (1024^2)
  gemm_wmma_bf16<128, 128, 64, 2, 4, false, float, float, 0,
                 false, false, true, false, false>
      <<<dim3(NV / 128, NV / 128, B_), 256, 0, stream>>>(
          T, (size_t)NV * NE, NE, T, (size_t)NV * NE, NE,
          dbf, nullptr, 2048,
          adjV, (size_t)NV * NV, Af, (size_t)NV * NV,
          nullptr, nullptr, 0, 0, 0, 0, NV, NV, NE);

  colmax_rcp_k<<<dim3(NV / 256, 1, B_), 256, 0, stream>>>(Af, rcm, NV);

  // XW3^T = (X1 @ W3)^T  (16 x 1024 f32 per batch)
  gemm_wmma_bf16<128, 16, 64, 8, 1, true, float, float, 1,
                 false, false, false, false, false>
      <<<dim3(C_ / 16, NV / 128, B_), 256, 0, stream>>>(
          X1, (size_t)NV * H_, H_, W3, 0, C_,
          nullptr, nullptr, 0, nullptr, 0, nullptr, 0,
          nullptr, XWT, (size_t)C_ * NV, NV, 0, 1, NV, C_, H_);

  // out = Af @ (XW3 * rcmax) + b3   (final, no relu; A via async-LDS)
  gemm_wmma_bf16<128, 16, 64, 8, 1, false, __bf16, float, 1,
                 true, false, false, true, false>
      <<<dim3(C_ / 16, NV / 128, B_), 256, 0, stream>>>(
          Af, (size_t)NV * NV, NV, XWT, (size_t)C_ * NV, NV,
          nullptr, rcm, 2048, nullptr, 0, nullptr, 0,
          b3, out, (size_t)NV * C_, C_, 0, 0, NV, C_, NV);
}
